// MVN_DDI_18021682774947
// MI455X (gfx1250) — matrix operand order repaired and checked by reading, not compiled, not run on hardware
//
#include <hip/hip_runtime.h>
#include <hip/hip_bf16.h>
#include <math.h>

// ---------------------------------------------------------------------------
// Problem constants (from reference)
// ---------------------------------------------------------------------------
#define NUM_GRAPHS 4096
#define NPG 24
#define EPG 48
#define LPG 144
#define T_ITER 4
#define HID 128
#define IN_DIM 70
#define EDGE_DIM 6
#define NNODES (NUM_GRAPHS * NPG)   // 98304
#define NEDGES (NUM_GRAPHS * EPG)   // 196608
#define NLG    (NUM_GRAPHS * LPG)   // 589824

// LDS row strides (halfs): multiples of 8 for 16B vector alignment,
// chosen so row-stride in dwords (52 / 68) rotates across the 64 banks.
#define LDA_X  104   // A tile with K=96 (padded)
#define LDA_H  136   // A tile with K=128 (padded)
#define LDK_W1 104   // transposed w_mlp  [128][96 ->104]
#define LDK_W  136   // transposed 128x128 weights [128][128 ->136]

typedef __attribute__((ext_vector_type(16))) _Float16 v16h;
typedef __attribute__((ext_vector_type(8)))  _Float16 v8h;
typedef __attribute__((ext_vector_type(8)))  float    v8f;

// ---------------------------------------------------------------------------
// WMMA fragment helpers (wave32, v_wmma_f32_16x16x32_f16)
// A tile: row-major f16 [16][lda]. Lane L: M=L&15, kh=L>>4.
//   Halves 0..7  <- K = kbase + kh*8 + 0..7      (one ds_load_b128)
//   Halves 8..15 <- K = kbase + 16 + kh*8 + 0..7 (one ds_load_b128)
// B tile: stored TRANSPOSED in LDS as [N][ldk]. Lane L: N=L&15, kh=L>>4.
//   Halves 0..7  <- K = kbase + kh*8 + 0..7; Halves 8..15 <- K = kbase + 16 + kh*8 + 0..7
// C/D: lane L: N=L&15, mh=L>>4; VGPR v holds M = v + 8*mh.
// ---------------------------------------------------------------------------
__device__ __forceinline__ v16h frag_a(const _Float16* A, int lda, int kbase) {
  int lane = threadIdx.x & 31;
  int m  = lane & 15;
  int kh = lane >> 4;
  const _Float16* p = A + m * lda + kbase + kh * 8;
  v8h lo = *(const v8h*)p;
  v8h hi = *(const v8h*)(p + 16);
  return __builtin_shufflevector(lo, hi, 0, 1, 2, 3, 4, 5, 6, 7,
                                 8, 9, 10, 11, 12, 13, 14, 15);
}

__device__ __forceinline__ v16h frag_bT(const _Float16* BT, int ldk, int kbase, int nbase) {
  int lane = threadIdx.x & 31;
  int n  = lane & 15;
  int kh = lane >> 4;
  const _Float16* p = BT + (nbase + n) * ldk + kbase + kh * 8;
  v8h lo = *(const v8h*)p;
  v8h hi = *(const v8h*)(p + 16);
  return __builtin_shufflevector(lo, hi, 0, 1, 2, 3, 4, 5, 6, 7,
                                 8, 9, 10, 11, 12, 13, 14, 15);
}

// ---------------------------------------------------------------------------
// Kernel 1: h = x @ w_mlp + b_mlp ; hu = h @ w_u ; hv = h @ w_v
// 256 threads (8 waves) per 16 node rows; each wave owns one 16-wide N tile.
// ---------------------------------------------------------------------------
__global__ void node_linears_kernel(const float* __restrict__ x,
                                    const float* __restrict__ w_mlp,
                                    const float* __restrict__ b_mlp,
                                    const float* __restrict__ w_u,
                                    const float* __restrict__ w_v,
                                    float* __restrict__ h,
                                    float* __restrict__ hu,
                                    float* __restrict__ hv) {
  extern __shared__ char smraw[];
  _Float16* sX   = (_Float16*)smraw;        // 16  x LDA_X   (K 70 -> 96 pad)
  _Float16* sWmT = sX   + 16 * LDA_X;       // 128 x LDK_W1  (w_mlp^T, zero pad)
  _Float16* sH   = sWmT + 128 * LDK_W1;     // 16  x LDA_H   (h as f16)
  _Float16* sWuT = sH   + 16 * LDA_H;       // 128 x LDK_W   (w_u^T)
  _Float16* sWvT = sWuT + 128 * LDK_W;      // 128 x LDK_W   (w_v^T)

  const int tid  = threadIdx.x;
  const int row0 = blockIdx.x * 16;

  for (int i = tid; i < 16 * 96; i += 256) {
    int r = i / 96, c = i - r * 96;
    float v = (c < IN_DIM) ? x[(long)(row0 + r) * IN_DIM + c] : 0.0f;
    sX[r * LDA_X + c] = (_Float16)v;
  }
  // transposed weight stage-in: global reads coalesced (i = k*128 + n)
  for (int i = tid; i < 96 * 128; i += 256) {
    int k = i >> 7, n = i & 127;
    sWmT[n * LDK_W1 + k] = (k < IN_DIM) ? (_Float16)w_mlp[i] : (_Float16)0.0f;
  }
  for (int i = tid; i < 128 * 128; i += 256) {
    int k = i >> 7, n = i & 127;
    sWuT[n * LDK_W + k] = (_Float16)w_u[i];
    sWvT[n * LDK_W + k] = (_Float16)w_v[i];
  }
  __syncthreads();

  const int lane = tid & 31;
  const int nb   = (tid >> 5) * 16;  // wave's N tile
  const int n    = lane & 15;
  const int mh   = lane >> 4;

  // ---- h = x @ w_mlp + b ----
  float bias = b_mlp[nb + n];
  v8f acc;
#pragma unroll
  for (int i = 0; i < 8; ++i) acc[i] = bias;
#pragma unroll
  for (int kb = 0; kb < 96; kb += 32) {
    v16h af = frag_a(sX, LDA_X, kb);
    v16h bf = frag_bT(sWmT, LDK_W1, kb, nb);
    acc = __builtin_amdgcn_wmma_f32_16x16x32_f16(false, af, false, bf,
                                                 (short)0, acc, false, false);
  }
#pragma unroll
  for (int r = 0; r < 8; ++r) {
    int m = r + 8 * mh;
    float val = acc[r];
    h[(long)(row0 + m) * HID + nb + n] = val;
    sH[m * LDA_H + nb + n] = (_Float16)val;
  }
  __syncthreads();

  // ---- hu = h @ w_u ; hv = h @ w_v ----
  v8f au, av;
#pragma unroll
  for (int i = 0; i < 8; ++i) { au[i] = 0.0f; av[i] = 0.0f; }
#pragma unroll
  for (int kb = 0; kb < 128; kb += 32) {
    v16h af = frag_a(sH, LDA_H, kb);
    v16h bu = frag_bT(sWuT, LDK_W, kb, nb);
    au = __builtin_amdgcn_wmma_f32_16x16x32_f16(false, af, false, bu,
                                                (short)0, au, false, false);
    v16h bv = frag_bT(sWvT, LDK_W, kb, nb);
    av = __builtin_amdgcn_wmma_f32_16x16x32_f16(false, af, false, bv,
                                                (short)0, av, false, false);
  }
#pragma unroll
  for (int r = 0; r < 8; ++r) {
    int m = r + 8 * mh;
    hu[(long)(row0 + m) * HID + nb + n] = au[r];
    hv[(long)(row0 + m) * HID + nb + n] = av[r];
  }
}

// ---------------------------------------------------------------------------
// Kernel 2: e0 = (hu[src] + hv[dst] + edge_attr @ w_edge) / 3
// One thread per (edge, 4 hid lanes): float4 gather + float4 store.
// ---------------------------------------------------------------------------
__global__ void edge_init_kernel(const float* __restrict__ hu,
                                 const float* __restrict__ hv,
                                 const float* __restrict__ edge_attr,
                                 const float* __restrict__ w_edge,
                                 const int*   __restrict__ edge_index,
                                 float* __restrict__ e0) {
  __shared__ float sWe[EDGE_DIM * HID];
  const int tid = threadIdx.x;
  for (int i = tid; i < EDGE_DIM * HID; i += 256) sWe[i] = w_edge[i];
  __syncthreads();

  long idx4 = (long)blockIdx.x * 256 + tid;      // < NEDGES * HID/4
  int e  = (int)(idx4 >> 5);                     // 32 float4 groups per edge
  int n4 = (int)(idx4 & 31) << 2;
  int s = edge_index[e];
  int d = edge_index[NEDGES + e];

  float4 u = *(const float4*)(hu + (long)s * HID + n4);
  float4 v = *(const float4*)(hv + (long)d * HID + n4);
  float4 r = make_float4(u.x + v.x, u.y + v.y, u.z + v.z, u.w + v.w);

  float ea[EDGE_DIM];
#pragma unroll
  for (int k = 0; k < EDGE_DIM; ++k) ea[k] = edge_attr[(long)e * EDGE_DIM + k];
#pragma unroll
  for (int k = 0; k < EDGE_DIM; ++k) {
    const float* w = &sWe[k * HID + n4];
    r.x += ea[k] * w[0]; r.y += ea[k] * w[1];
    r.z += ea[k] * w[2]; r.w += ea[k] * w[3];
  }
  const float third = 1.0f / 3.0f;
  r.x *= third; r.y *= third; r.z *= third; r.w *= third;
  *(float4*)(e0 + idx4 * 4) = r;
}

// ---------------------------------------------------------------------------
// Kernel 3: per-graph iterative line-graph message passing + attention pool,
// all LDS-resident. One block of 256 threads (8 waves) per graph.
// ---------------------------------------------------------------------------
__global__ void graph_loop_kernel(const float* __restrict__ e0g,
                                  const float* __restrict__ hg,
                                  const int*   __restrict__ edge_index,
                                  const int*   __restrict__ lg_edge_index,
                                  const float* __restrict__ w_rel,
                                  const float* __restrict__ b_rel,
                                  const float* __restrict__ w_root,
                                  const float* __restrict__ a_w,
                                  const float* __restrict__ a_bias,
                                  const float* __restrict__ w_gout,
                                  const float* __restrict__ b_gout,
                                  float* __restrict__ xo_pre) {
  extern __shared__ float smf[];
  float* sE0    = smf;                          // 48*128
  float* sHist  = sE0 + EPG * HID;              // 4 * 48*128
  float* sNode  = sHist + T_ITER * EPG * HID;   // 24*128
  float* sWrel  = sNode + NPG * HID;            // 128
  float* sWroot = sWrel + HID;                  // 128
  float* sPr    = sWroot + HID;                 // 48
  float* sQr    = sPr + EPG;                    // 48
  float* sXc    = sQr + EPG;                    // 48
  float* sSc    = sXc + EPG;                    // 48
  float* sG     = sSc + EPG;                    // 128
  float* sGout  = sG + HID;                     // 4*128
  float* sRed   = sGout + T_ITER * HID;         // 8
  float* sScore = sRed + 8;                     // 4
  float* sWt    = sScore + 4;                   // 4
  int*   sLsrc  = (int*)(sWt + 4);              // 144
  int*   sLdst  = sLsrc + LPG;                  // 144
  int*   sDst   = sLdst + LPG;                  // 48

  const int tid   = threadIdx.x;
  const int lane  = tid & 31;
  const int wid   = tid >> 5;
  const int g     = blockIdx.x;
  const int ebase = g * EPG;
  const int nbase = g * NPG;

  // warm L2 for the shared w_gout matrix (global_prefetch_b8)
  for (int i = tid * 32; i < HID * HID; i += 256 * 32)
    __builtin_prefetch(&w_gout[i], 0, 3);

  for (int i = tid; i < LPG; i += 256) {
    sLsrc[i] = lg_edge_index[(long)g * LPG + i] - ebase;
    sLdst[i] = lg_edge_index[(long)NLG + (long)g * LPG + i] - ebase;
  }
  if (tid < EPG) sDst[tid] = edge_index[(long)NEDGES + ebase + tid] - nbase;
  for (int i = tid; i < HID; i += 256) { sWrel[i] = w_rel[i]; sWroot[i] = w_root[i]; }
  for (int i = tid; i < EPG * HID / 4; i += 256)
    ((float4*)sE0)[i] = ((const float4*)(e0g + (long)ebase * HID))[i];
  __syncthreads();

  const float brel = b_rel[0];

  for (int t = 0; t < T_ITER; ++t) {
    float* outPrev = (t == 0) ? sE0 : (sHist + (t - 1) * EPG * HID);
    float* outCur  = sHist + t * EPG * HID;

    for (int i = tid; i < EPG * HID; i += 256) outCur[i] = sE0[i];
    __syncthreads();

    // out_t = e0 + segment_sum(out_prev[lsrc], ldst)   (ds_add_f32 scatter)
    for (int i = tid; i < LPG * HID; i += 256) {
      int l = i >> 7, n = i & 127;
      atomicAdd(&outCur[sLdst[l] * HID + n], outPrev[sLsrc[l] * HID + n]);
    }
    __syncthreads();

    // pr[e] = dot(out_t[e], w_rel), qr[e] = dot(out_t[e], w_root)
    // one edge per wave iteration; butterfly reduction, no LDS atomics
    for (int e = wid; e < EPG; e += 8) {
      float pp = 0.0f, qq = 0.0f;
      for (int n = lane; n < HID; n += 32) {
        float o = outCur[e * HID + n];
        pp += o * sWrel[n];
        qq += o * sWroot[n];
      }
#pragma unroll
      for (int off = 16; off > 0; off >>= 1) {
        pp += __shfl_xor(pp, off);
        qq += __shfl_xor(qq, off);
      }
      if (lane == 0) { sPr[e] = pp; sQr[e] = qq; }
    }
    __syncthreads();

    // xc[e] = b_rel + qr[e] + sum_{l: ldst=e} pr[lsrc]
    if (tid < EPG) sXc[tid] = brel + sQr[tid];
    __syncthreads();
    for (int l = tid; l < LPG; l += 256) atomicAdd(&sXc[sLdst[l]], sPr[sLsrc[l]]);
    __syncthreads();

    // segment softmax over this graph's 48 edges
    if (tid == 0) {
      float m = sXc[0];
      for (int e = 1; e < EPG; ++e) m = fmaxf(m, sXc[e]);
      sRed[0] = m;
    }
    __syncthreads();
    if (tid < EPG) sSc[tid] = __expf(sXc[tid] - sRed[0]);
    __syncthreads();
    if (tid == 0) {
      float s = 0.0f;
      for (int e = 0; e < EPG; ++e) s += sSc[e];
      sRed[1] = 1.0f / s;
    }
    __syncthreads();

    // g[n] = sum_e out_t[e][n] * softmax(e)
    if (tid < HID) {
      float gn = 0.0f;
      for (int e = 0; e < EPG; ++e) gn += outCur[e * HID + tid] * sSc[e];
      sG[tid] = gn * sRed[1];
    }
    __syncthreads();

    // gout_t = tanh(g @ w_gout + b_gout)  (w_gout streams from L2)
    if (tid < HID) {
      float acc = b_gout[tid];
      for (int k = 0; k < HID; ++k) acc += sG[k] * w_gout[k * HID + tid];
      sGout[t * HID + tid] = tanhf(acc);
    }
    __syncthreads();
  }

  // temporal attention over T iterations
  if (tid < T_ITER) {
    float s = a_bias[tid];
    for (int hh = 0; hh < HID; ++hh) s += sGout[tid * HID + hh] * a_w[hh * T_ITER + tid];
    sScore[tid] = s;
  }
  __syncthreads();
  if (tid == 0) {
    float m = fmaxf(fmaxf(sScore[0], sScore[1]), fmaxf(sScore[2], sScore[3]));
    float ex0 = __expf(sScore[0] - m), ex1 = __expf(sScore[1] - m);
    float ex2 = __expf(sScore[2] - m), ex3 = __expf(sScore[3] - m);
    float inv = 1.0f / (ex0 + ex1 + ex2 + ex3);
    sWt[0] = ex0 * inv; sWt[1] = ex1 * inv; sWt[2] = ex2 * inv; sWt[3] = ex3 * inv;
  }
  for (int i = tid; i < NPG * HID; i += 256) sNode[i] = 0.0f;
  __syncthreads();

  // out_final = sum_t wt[t]*out_t ; scatter-add to dst nodes
  for (int i = tid; i < EPG * HID; i += 256) {
    int e = i >> 7, n = i & 127;
    float of = sWt[0] * sHist[i]
             + sWt[1] * sHist[EPG * HID + i]
             + sWt[2] * sHist[2 * EPG * HID + i]
             + sWt[3] * sHist[3 * EPG * HID + i];
    atomicAdd(&sNode[sDst[e] * HID + n], of);
  }
  __syncthreads();

  for (int i = tid; i < NPG * HID / 4; i += 256) {
    float4 hv4 = ((const float4*)(hg + (long)nbase * HID))[i];
    float4 nd  = ((const float4*)sNode)[i];
    ((float4*)(xo_pre + (long)nbase * HID))[i] =
        make_float4(hv4.x + nd.x, hv4.y + nd.y, hv4.z + nd.z, hv4.w + nd.w);
  }
}

// ---------------------------------------------------------------------------
// Kernel 4: d_out = xo_pre @ w_block + b_block  (WMMA)
// ---------------------------------------------------------------------------
__global__ void out_gemm_kernel(const float* __restrict__ xo_pre,
                                const float* __restrict__ w_block,
                                const float* __restrict__ b_block,
                                float* __restrict__ out) {
  extern __shared__ char smraw[];
  _Float16* sA  = (_Float16*)smraw;   // 16  x LDA_H
  _Float16* sWT = sA + 16 * LDA_H;    // 128 x LDK_W (transposed)

  const int tid  = threadIdx.x;
  const int row0 = blockIdx.x * 16;

  for (int i = tid; i < 16 * 128; i += 256) {
    int r = i >> 7, c = i & 127;
    sA[r * LDA_H + c] = (_Float16)xo_pre[(long)row0 * HID + i];
  }
  for (int i = tid; i < 128 * 128; i += 256) {
    int k = i >> 7, n = i & 127;
    sWT[n * LDK_W + k] = (_Float16)w_block[i];
  }
  __syncthreads();

  const int lane = tid & 31;
  const int nb   = (tid >> 5) * 16;
  const int n    = lane & 15;
  const int mh   = lane >> 4;

  float bias = b_block[nb + n];
  v8f acc;
#pragma unroll
  for (int i = 0; i < 8; ++i) acc[i] = bias;
#pragma unroll
  for (int kb = 0; kb < 128; kb += 32) {
    v16h af = frag_a(sA, LDA_H, kb);
    v16h bf = frag_bT(sWT, LDK_W, kb, nb);
    acc = __builtin_amdgcn_wmma_f32_16x16x32_f16(false, af, false, bf,
                                                 (short)0, acc, false, false);
  }
#pragma unroll
  for (int r = 0; r < 8; ++r) {
    int m = r + 8 * mh;
    out[(long)(row0 + m) * HID + nb + n] = acc[r];
  }
}

// ---------------------------------------------------------------------------
// Host launcher
// ---------------------------------------------------------------------------
extern "C" void kernel_launch(void* const* d_in, const int* in_sizes, int n_in,
                              void* d_out, int out_size, void* d_ws, size_t ws_size,
                              hipStream_t stream) {
  (void)in_sizes; (void)n_in; (void)out_size; (void)ws_size;

  const float* x             = (const float*)d_in[0];
  const float* edge_attr     = (const float*)d_in[1];
  const int*   edge_index    = (const int*)d_in[2];
  const int*   lg_edge_index = (const int*)d_in[3];
  /* d_in[4] = edge_batch (unused: per-graph blocks) */
  const float* w_mlp   = (const float*)d_in[5];
  const float* b_mlp   = (const float*)d_in[6];
  const float* w_u     = (const float*)d_in[7];
  const float* w_v     = (const float*)d_in[8];
  const float* w_edge  = (const float*)d_in[9];
  const float* w_rel   = (const float*)d_in[10];
  const float* b_rel   = (const float*)d_in[11];
  const float* w_root  = (const float*)d_in[12];
  const float* a_w     = (const float*)d_in[13];
  const float* a_bias  = (const float*)d_in[14];
  const float* w_gout  = (const float*)d_in[15];
  const float* b_gout  = (const float*)d_in[16];
  const float* w_block = (const float*)d_in[17];
  const float* b_block = (const float*)d_in[18];

  float* ws     = (float*)d_ws;
  float* h      = ws;                              // NNODES*HID
  float* hu     = h  + (size_t)NNODES * HID;       // NNODES*HID
  float* hv     = hu + (size_t)NNODES * HID;       // NNODES*HID
  float* e0     = hv + (size_t)NNODES * HID;       // NEDGES*HID
  float* xo_pre = e0 + (size_t)NEDGES * HID;       // NNODES*HID

  // Kernel 1: node-side GEMMs (WMMA)
  size_t lds1 = (size_t)(16 * LDA_X + 128 * LDK_W1 + 16 * LDA_H + 2 * 128 * LDK_W)
              * sizeof(_Float16);
  node_linears_kernel<<<NNODES / 16, 256, lds1, stream>>>(x, w_mlp, b_mlp, w_u, w_v,
                                                          h, hu, hv);

  // Kernel 2: directed-edge init (float4 per thread)
  edge_init_kernel<<<(NEDGES * (HID / 4)) / 256, 256, 0, stream>>>(hu, hv, edge_attr,
                                                                   w_edge, edge_index, e0);

  // Kernel 3: per-graph LDS-resident iterative loop
  size_t lds3 = (size_t)((EPG * HID) + (T_ITER * EPG * HID) + (NPG * HID)
                         + 2 * HID + 4 * EPG + HID + T_ITER * HID + 16) * sizeof(float)
              + (size_t)(2 * LPG + EPG) * sizeof(int);
  graph_loop_kernel<<<NUM_GRAPHS, 256, lds3, stream>>>(e0, h, edge_index, lg_edge_index,
                                                       w_rel, b_rel, w_root, a_w, a_bias,
                                                       w_gout, b_gout, xo_pre);

  // Kernel 4: final linear (WMMA)
  size_t lds4 = (size_t)(16 * LDA_H + 128 * LDK_W) * sizeof(_Float16);
  out_gemm_kernel<<<NNODES / 16, 256, lds4, stream>>>(xo_pre, w_block, b_block,
                                                      (float*)d_out);
}
